// LSTMCell_62964220559530
// MI455X (gfx1250) — compile-verified, matrix operand order checked by reading, not run on hardware
//
#include <hip/hip_runtime.h>
#include <hip/hip_bf16.h>
#include <math.h>

// Problem dims (match the reference)
#define B_DIM 8192
#define IN_DIM 1024
#define H_DIM 1024
#define K_DIM (IN_DIM + H_DIM)   // 2048
#define N_DIM (4 * H_DIM)        // 4096

typedef __attribute__((ext_vector_type(16))) __bf16 v16bf;
typedef __attribute__((ext_vector_type(8)))  __bf16 v8bf;
typedef __attribute__((ext_vector_type(8)))  float  v8f;

// float -> bf16, round-to-nearest-even
__device__ __forceinline__ unsigned int f2bf(float f) {
    unsigned int u = __float_as_uint(f);
    unsigned int r = u + 0x7FFFu + ((u >> 16) & 1u);
    return (r >> 16) & 0xFFFFu;
}

__device__ __forceinline__ float sigmoidf_fast(float x) {
    return 1.0f / (1.0f + __expf(-x));
}

// ---------------------------------------------------------------------------
// Pack [x | h] rows into bf16 [B, 2048], row-major. 8 elements per thread.
// ---------------------------------------------------------------------------
__global__ __launch_bounds__(256) void pack_xh_kernel(
    const float* __restrict__ x, const float* __restrict__ h,
    unsigned short* __restrict__ xh)
{
    size_t e   = ((size_t)blockIdx.x * blockDim.x + threadIdx.x) * 8; // < B*K
    size_t row = e >> 11;              // / 2048
    int    k   = (int)(e & (K_DIM - 1));
    const float* src = (k < IN_DIM) ? (x + row * IN_DIM + k)
                                    : (h + row * H_DIM + (k - IN_DIM));
    float4 f0 = *(const float4*)(src);
    float4 f1 = *(const float4*)(src + 4);
    uint4 o;
    o.x = f2bf(f0.x) | (f2bf(f0.y) << 16);
    o.y = f2bf(f0.z) | (f2bf(f0.w) << 16);
    o.z = f2bf(f1.x) | (f2bf(f1.y) << 16);
    o.w = f2bf(f1.z) | (f2bf(f1.w) << 16);
    *(uint4*)(xh + e) = o;
}

// ---------------------------------------------------------------------------
// Tiled transpose + bf16 pack: src [1024, 4096] f32 -> dst [4096, 2048] bf16
// at column offset koff (0 for Wx, 1024 for Wh). 32x32 tiles via LDS.
// ---------------------------------------------------------------------------
__global__ __launch_bounds__(256) void pack_w_kernel(
    const float* __restrict__ src, unsigned short* __restrict__ dst, int koff)
{
    __shared__ unsigned short tile[32][33];   // +1 pad to dodge bank conflicts
    const int nbase = blockIdx.x * 32;        // column of src == row of dst
    const int kbase = blockIdx.y * 32;        // row of src == column of dst
    const int tx = threadIdx.x, ty = threadIdx.y;  // (32, 8)

    #pragma unroll
    for (int i = ty; i < 32; i += 8) {
        float v = src[(size_t)(kbase + i) * N_DIM + nbase + tx];
        tile[i][tx] = (unsigned short)f2bf(v);
    }
    __syncthreads();
    #pragma unroll
    for (int i = ty; i < 32; i += 8) {
        dst[(size_t)(nbase + i) * K_DIM + koff + kbase + tx] = tile[tx][i];
    }
}

// ---------------------------------------------------------------------------
// Fused LSTM GEMM + epilogue.
// Block: 512 threads (16 wave32) arranged 8(M) x 2(N).
// Wave tile : 32 rows x 16 units x 4 gates.
// Block tile: 256 rows x 32 units x 4 gates. Grid: (H/32, B/256).
// K-step 64 (two 32-K sub-chunks per LDS buffer), double-buffered via
// GLOBAL_LOAD_ASYNC_TO_LDS_B128 (ASYNCcnt), one barrier per K-step.
// ---------------------------------------------------------------------------
__global__ __launch_bounds__(512) void lstm_gemm_kernel(
    const unsigned short* __restrict__ xh,   // [B, 2048] bf16
    const unsigned short* __restrict__ Wt,   // [4096, 2048] bf16 (N-major)
    const float* __restrict__ c_old,         // [B, H]
    const float* __restrict__ bx,            // [4H]
    const float* __restrict__ bh,            // [4H]
    float* __restrict__ out)                 // [3, B, H] : o, h_new, c_new
{
    // 2 buffers x (4 gates x 32 cols x 64 K) bf16 = 2 x 16KB = 32KB
    __shared__ unsigned short ldsB[2 * 4 * 32 * 64];

    const int tid   = threadIdx.x;
    const int wave  = tid >> 5;
    const int lane  = tid & 31;
    const int mwave = wave & 7;           // 0..7  : row sub-tile
    const int nwave = wave >> 3;          // 0..1  : col sub-tile
    const int n0    = blockIdx.x * 32;    // hidden-unit tile base (within H)
    const int m0    = blockIdx.y * 256;   // row tile base

    const int khalf = lane >> 4;          // 0: K {0..7,16..23}, 1: K {8..15,24..31}
    const int nl    = lane & 15;
    const int nn    = nwave * 16 + nl;    // column within block tile (0..31)

    // Accumulators: two 16-row subtiles x 4 gates, 16x16 f32 each
    v8f acc[2][4];
    #pragma unroll
    for (int t = 0; t < 2; ++t)
        #pragma unroll
        for (int g = 0; g < 4; ++g) acc[t][g] = (v8f){};

    // A fragment sources: rows m0 + mwave*32 + nl (+16 for subtile 1)
    const unsigned short* aptr0 =
        xh + (size_t)(m0 + mwave * 32 + nl) * K_DIM + khalf * 8;
    const unsigned short* aptr1 = aptr0 + (size_t)16 * K_DIM;

    // Cooperative async LDS fill: 512 threads x 2 x 16B = 16KB per K-step.
    // Entry idx -> gate g (idx>>8), col n ((idx&255)>>3), chunk ch (idx&7).
    // Thread handles idx = tid (g in 0..1) and idx = tid+512 (g in 2..3).
    const int g_f  = tid >> 8;            // 0..1
    const int r_f  = tid & 255;
    const int n_f  = r_f >> 3;            // 0..31
    const int ch_f = r_f & 7;             // 0..7 (8 elems each -> 64 K)
    const unsigned fill_off0 =            // byte offset inside a buffer
        (unsigned)(g_f * 2048 + n_f * 64 + ch_f * 8) * 2u;
    const unsigned fill_off1 = fill_off0 + 8192u;       // gate g_f + 2
    unsigned voff0 =                       // byte offset into Wt (k0 = 0)
        (unsigned)(((size_t)(g_f * H_DIM + n0 + n_f) * K_DIM + ch_f * 8) * 2u);
    unsigned voff1 = voff0 + (unsigned)(2u * H_DIM * K_DIM * 2u);
    const unsigned ldsBase = (unsigned)(size_t)(void*)ldsB;  // LDS byte address

    // Prologue: fill buffer 0 for k0 = 0
    asm volatile("global_load_async_to_lds_b128 %0, %1, %2"
                 :: "v"(ldsBase + fill_off0), "v"(voff0), "s"(Wt) : "memory");
    asm volatile("global_load_async_to_lds_b128 %0, %1, %2"
                 :: "v"(ldsBase + fill_off1), "v"(voff1), "s"(Wt) : "memory");

    for (int k0 = 0; k0 < K_DIM; k0 += 64) {
        const int buf = (k0 >> 6) & 1;
        const unsigned bufbase = (unsigned)buf * 8192u;   // uint16 index

        // my fills done; then all fills done & all reads of buf^1 done
        asm volatile("s_wait_asynccnt 0x0" ::: "memory");
        __syncthreads();

        // kick off next tile's fills into the other buffer (overlaps compute)
        if (k0 + 64 < K_DIM) {
            voff0 += 128;
            voff1 += 128;
            const unsigned dstb = ldsBase + (unsigned)(buf ^ 1) * 16384u;
            asm volatile("global_load_async_to_lds_b128 %0, %1, %2"
                         :: "v"(dstb + fill_off0), "v"(voff0), "s"(Wt) : "memory");
            asm volatile("global_load_async_to_lds_b128 %0, %1, %2"
                         :: "v"(dstb + fill_off1), "v"(voff1), "s"(Wt) : "memory");
        }

        #pragma unroll
        for (int kc = 0; kc < 2; ++kc) {
            const int kk = k0 + kc * 32;

            // A fragments (16x32 bf16): two contiguous 16B chunks per lane
            v8bf a0lo = *(const v8bf*)(aptr0 + kk);
            v8bf a0hi = *(const v8bf*)(aptr0 + kk + 16);
            v8bf a1lo = *(const v8bf*)(aptr1 + kk);
            v8bf a1hi = *(const v8bf*)(aptr1 + kk + 16);
            v16bf a0 = __builtin_shufflevector(a0lo, a0hi,
                         0, 1, 2, 3, 4, 5, 6, 7, 8, 9, 10, 11, 12, 13, 14, 15);
            v16bf a1 = __builtin_shufflevector(a1lo, a1hi,
                         0, 1, 2, 3, 4, 5, 6, 7, 8, 9, 10, 11, 12, 13, 14, 15);

            // Preload all four gate B-fragments, then burst 8 WMMAs
            v16bf bfrag[4];
            #pragma unroll
            for (int g = 0; g < 4; ++g) {
                const unsigned short* bp =
                    &ldsB[bufbase + g * 2048 + nn * 64 + kc * 32 + khalf * 8];
                v8bf blo = *(const v8bf*)bp;
                v8bf bhi = *(const v8bf*)(bp + 16);
                bfrag[g] = __builtin_shufflevector(blo, bhi,
                         0, 1, 2, 3, 4, 5, 6, 7, 8, 9, 10, 11, 12, 13, 14, 15);
            }
            #pragma unroll
            for (int g = 0; g < 4; ++g) {
                acc[0][g] = __builtin_amdgcn_wmma_f32_16x16x32_bf16(
                                false, a0, false, bfrag[g], (short)0, acc[0][g],
                                false, false);
                acc[1][g] = __builtin_amdgcn_wmma_f32_16x16x32_bf16(
                                false, a1, false, bfrag[g], (short)0, acc[1][g],
                                false, false);
            }
        }
    }

    // ---------------- Epilogue: biases + activations + LSTM update ----------
    const int col = n0 + nn;
    float bsum[4];
    #pragma unroll
    for (int g = 0; g < 4; ++g)
        bsum[g] = bx[g * H_DIM + col] + bh[g * H_DIM + col];

    const size_t BH = (size_t)B_DIM * H_DIM;

    #pragma unroll
    for (int t = 0; t < 2; ++t) {
        const int rowbase = m0 + mwave * 32 + t * 16;
        #pragma unroll
        for (int j = 0; j < 8; ++j) {
            // C/D layout: VGPR j, lanes 0-15 -> M=j, lanes 16-31 -> M=j+8
            const int row = rowbase + j + khalf * 8;
            const size_t idx = (size_t)row * H_DIM + col;

            float iv = sigmoidf_fast(acc[t][0][j] + bsum[0]);
            float fv = sigmoidf_fast(acc[t][1][j] + bsum[1]);
            float gv = tanhf(acc[t][2][j] + bsum[2]);
            float ov = sigmoidf_fast(acc[t][3][j] + bsum[3]);

            float cn = fv * c_old[idx] + iv * gv;
            float hn = ov * tanhf(cn);

            out[idx]          = ov;  // out_gate
            out[BH + idx]     = hn;  // hidden_state
            out[2 * BH + idx] = cn;  // cell_state
        }
    }
}

// ---------------------------------------------------------------------------
// Host-side launcher (graph-capture safe: only kernel launches on `stream`).
// Workspace layout: [0, 32MB) xh bf16 ; [32MB, 48MB) Wt bf16.
// ---------------------------------------------------------------------------
extern "C" void kernel_launch(void* const* d_in, const int* in_sizes, int n_in,
                              void* d_out, int out_size, void* d_ws, size_t ws_size,
                              hipStream_t stream) {
    const float* x  = (const float*)d_in[0];
    const float* h  = (const float*)d_in[1];
    const float* c  = (const float*)d_in[2];
    const float* Wx = (const float*)d_in[3];
    const float* Wh = (const float*)d_in[4];
    const float* bx = (const float*)d_in[5];
    const float* bh = (const float*)d_in[6];
    float* out = (float*)d_out;

    unsigned char* ws = (unsigned char*)d_ws;
    unsigned short* xh_bf = (unsigned short*)ws;                               // 32MB
    unsigned short* Wt_bf = (unsigned short*)(ws + (size_t)B_DIM * K_DIM * 2); // 16MB

    // Pack inputs to bf16
    {
        size_t total = (size_t)B_DIM * K_DIM / 8;      // 2M threads
        pack_xh_kernel<<<dim3((unsigned)(total / 256)), dim3(256), 0, stream>>>(x, h, xh_bf);
    }
    {
        dim3 grid(N_DIM / 32, IN_DIM / 32);            // 128 x 32
        dim3 blk(32, 8);
        pack_w_kernel<<<grid, blk, 0, stream>>>(Wx, Wt_bf, 0);
        pack_w_kernel<<<grid, blk, 0, stream>>>(Wh, Wt_bf, IN_DIM);
    }

    // Fused GEMM + LSTM epilogue
    dim3 grid(H_DIM / 32, B_DIM / 256);                // 32 x 32 blocks
    lstm_gemm_kernel<<<grid, dim3(512), 0, stream>>>(xh_bf, Wt_bf, c, bx, bh, out);
}
